// dot_attention_73400991088959
// MI455X (gfx1250) — matrix operand order repaired and checked by reading, not compiled, not run on hardware
//
#include <hip/hip_runtime.h>

typedef _Float16 f16_t;
typedef __attribute__((ext_vector_type(16))) _Float16 v16h;
typedef __attribute__((ext_vector_type(8)))  _Float16 v8h;
typedef __attribute__((ext_vector_type(8)))  float    v8f;
typedef __attribute__((ext_vector_type(4)))  float    v4f;

#define BATCH 16
#define SEQ   2048
#define DHEAD 64
#define PSTR  2052              // padded LDS row stride (floats): 2052 % 64 == 4
#define SCALE 0.125f
#define NWAVES 4
#define NTHREADS (NWAVES * 32)

// ---------------- fragment helpers ----------------

// A-matrix fragment (16x32 f16) from row-major fp32 data, scaled.
// 16-bit A layout (matches B): lane-group g=lane/16, element e:
//   K = koff + g*16 + e
__device__ __forceinline__ v16h fragA_f32(const float* __restrict__ rowptr, int g, float scale) {
  v16h f;
  const float* p0 = rowptr + g * 16;
  const float* p1 = rowptr + g * 16 + 8;
#pragma unroll
  for (int j = 0; j < 8; ++j) f[j] = (_Float16)(p0[j] * scale);
#pragma unroll
  for (int j = 0; j < 8; ++j) f[8 + j] = (_Float16)(p1[j] * scale);
  return f;
}

// B-matrix fragment (32x16 f16), column held by lane%16, element e -> K = g*16 + e.
__device__ __forceinline__ v16h fragB_f32(const float* __restrict__ p) {
  v16h f;
#pragma unroll
  for (int j = 0; j < 16; ++j) f[j] = (_Float16)p[j];
  return f;
}

// Same fragment from pre-converted f16 (32B-aligned contiguous).
__device__ __forceinline__ v16h fragB_f16(const f16_t* __restrict__ p) {
  return *(const v16h*)p;
}

// ---------------- prep kernels ----------------

// Elementwise fp32 -> f16 (K matrix), 4 elems/thread. RT stores: reused from L2.
__global__ void cvt_f16_kernel(const float* __restrict__ in, f16_t* __restrict__ out) {
  int i = blockIdx.x * blockDim.x + threadIdx.x;   // i indexes float4 chunks
  float4 x = ((const float4*)in)[i];
  typedef __attribute__((ext_vector_type(4))) _Float16 v4h;
  v4h h = {(_Float16)x.x, (_Float16)x.y, (_Float16)x.z, (_Float16)x.w};
  ((v4h*)out)[i] = h;
}

// V[b][k][n] fp32  ->  Vt[b][n][k] f16 (K-major), 64x64 tiles via LDS.
__global__ void vtrans_kernel(const float* __restrict__ V, f16_t* __restrict__ Vt) {
  __shared__ f16_t t[64][72];                       // pad: 144B row stride (16B aligned)
  const int b  = blockIdx.x >> 5;                   // 32 k-tiles per batch
  const int kt = blockIdx.x & 31;
  const float* src = V + ((size_t)b * SEQ + (size_t)kt * 64) * DHEAD;
  for (int i = threadIdx.x; i < 64 * 64 / 4; i += 256) {
    const int row = i >> 4;                         // k within tile
    const int c4  = i & 15;                         // float4 within row
    float4 x = *(const float4*)(src + row * DHEAD + c4 * 4);
    t[c4 * 4 + 0][row] = (_Float16)x.x;
    t[c4 * 4 + 1][row] = (_Float16)x.y;
    t[c4 * 4 + 2][row] = (_Float16)x.z;
    t[c4 * 4 + 3][row] = (_Float16)x.w;
  }
  __syncthreads();
  f16_t* dst = Vt + (size_t)b * DHEAD * SEQ + (size_t)kt * 64;
  for (int i = threadIdx.x; i < 64 * 64 / 8; i += 256) {
    const int n  = i >> 3;
    const int c8 = i & 7;
    *(v8h*)(dst + (size_t)n * SEQ + c8 * 8) = *(const v8h*)&t[n][c8 * 8];
  }
}

// ---------------- main fused attention kernel ----------------

template <bool PREP>
__global__ __launch_bounds__(NTHREADS) void attn_fused_kernel(
    const float* __restrict__ Q, const float* __restrict__ K,
    const float* __restrict__ V, const f16_t* __restrict__ Kh,
    const f16_t* __restrict__ Vt, float* __restrict__ out) {
  extern __shared__ float smem[];
  float* P        = smem;                 // [16][PSTR] scores -> exp probs
  float* red      = P + 16 * PSTR;        // [16][8]
  float* rowmax   = red + 128;            // [16]
  float* rowscale = rowmax + 16;          // [16] = 1/rowsum

  const int tid  = threadIdx.x;
  const int wave = tid >> 5;
  const int lane = tid & 31;
  const int g    = lane >> 4;
  const int lr   = lane & 15;

  const int b     = blockIdx.x >> 7;
  const int qbase = (blockIdx.x & 127) << 4;

  const float* Qb = Q + ((size_t)b * SEQ + qbase) * DHEAD;
  const float* Kb = K + (size_t)b * SEQ * DHEAD;
  const float* Vb = V + (size_t)b * SEQ * DHEAD;

  // ---- Phase 1: scores = (Q*SCALE) * K^T  (f16 WMMA, f32 accumulate) ----
  v16h a0 = fragA_f32(Qb + lr * DHEAD + 0,  g, SCALE);
  v16h a1 = fragA_f32(Qb + lr * DHEAD + 32, g, SCALE);

  for (int nt = wave; nt < SEQ / 16; nt += NWAVES) {
    v16h b0, b1;
    if (PREP) {
      const f16_t* krow = Kh + ((size_t)b * SEQ + nt * 16 + lr) * DHEAD;
      b0 = fragB_f16(krow + 0  + g * 16);
      b1 = fragB_f16(krow + 32 + g * 16);
    } else {
      const float* krow = Kb + (size_t)(nt * 16 + lr) * DHEAD;
      b0 = fragB_f32(krow + 0  + g * 16);
      b1 = fragB_f32(krow + 32 + g * 16);
    }
    v8f c = {};
    c = __builtin_amdgcn_wmma_f32_16x16x32_f16(false, a0, false, b0, (short)0, c, false, false);
    c = __builtin_amdgcn_wmma_f32_16x16x32_f16(false, a1, false, b1, (short)0, c, false, false);
#pragma unroll
    for (int r = 0; r < 8; ++r)
      P[(r + g * 8) * PSTR + nt * 16 + lr] = c[r];
  }
  __syncthreads();

  // ---- Softmax (fp32, float4-vectorized): 8 threads/row, 256 cols each ----
  {
    const int row = tid >> 3;
    const int seg = tid & 7;
    float4* pr4 = (float4*)(P + row * PSTR + seg * 256);
    float m = -3.0e38f;
    for (int i = 0; i < 64; ++i) {
      float4 x = pr4[i];
      m = fmaxf(m, fmaxf(fmaxf(x.x, x.y), fmaxf(x.z, x.w)));
    }
    red[row * 8 + seg] = m;
    __syncthreads();
    if (tid < 16) {
      float mm = red[tid * 8];
#pragma unroll
      for (int j = 1; j < 8; ++j) mm = fmaxf(mm, red[tid * 8 + j]);
      rowmax[tid] = mm;
    }
    __syncthreads();
    const float mm = rowmax[row];
    float s = 0.0f;
    for (int i = 0; i < 64; ++i) {
      float4 x = pr4[i];
      x.x = __expf(x.x - mm); x.y = __expf(x.y - mm);
      x.z = __expf(x.z - mm); x.w = __expf(x.w - mm);
      pr4[i] = x;                      // keep unnormalized exp in LDS
      s += (x.x + x.y) + (x.z + x.w);
    }
    red[row * 8 + seg] = s;
    __syncthreads();
    if (tid < 16) {
      float ss = 0.0f;
#pragma unroll
      for (int j = 0; j < 8; ++j) ss += red[tid * 8 + j];
      rowscale[tid] = 1.0f / ss;
    }
    __syncthreads();
  }

  // ---- Write attention matrix: streaming 268 MB, non-temporal stores so the
  // ---- hot Kh/Vt working set stays L2-resident ----
  {
    float* attn = out + (size_t)BATCH * SEQ * DHEAD + ((size_t)b * SEQ + qbase) * SEQ;
    for (int idx = tid; idx < 16 * SEQ / 4; idx += NTHREADS) {
      const int r  = idx >> 9;                     // 512 float4 per row
      const int c4 = idx & 511;
      v4f x = *(const v4f*)(P + r * PSTR + c4 * 4);
      const float s = rowscale[r];
      x *= s;
      __builtin_nontemporal_store(x, (v4f*)(attn + (size_t)r * SEQ + c4 * 4));
    }
  }

  // ---- Phase 2: context = softmax(P) * V ; wave owns cols [wave*16,+16) ----
  {
    v8f c0 = {}, c1 = {};
    const int ncol = wave * 16 + lr;
    const float* vp  = Vb + ncol;                             // non-prep path
    const f16_t* vtp = PREP ? (Vt + ((size_t)b * DHEAD + ncol) * SEQ) : (const f16_t*)0;
    for (int kt = 0; kt < SEQ / 32; kt += 2) {
#pragma unroll
      for (int u = 0; u < 2; ++u) {
        const int kk = kt + u;
        v16h a = fragA_f32(P + lr * PSTR + kk * 32, g, 1.0f);
        v16h bf;
        if (PREP) {
          bf = fragB_f16(vtp + kk * 32 + g * 16);
        } else {
          const int kbase = kk * 32 + g * 16;
#pragma unroll
          for (int e = 0; e < 16; ++e)
            bf[e] = (_Float16)vp[(size_t)(kbase + e) * DHEAD];
        }
        if (u == 0)
          c0 = __builtin_amdgcn_wmma_f32_16x16x32_f16(false, a, false, bf, (short)0, c0, false, false);
        else
          c1 = __builtin_amdgcn_wmma_f32_16x16x32_f16(false, a, false, bf, (short)0, c1, false, false);
      }
    }
    v8f c = c0 + c1;
#pragma unroll
    for (int r = 0; r < 8; ++r) {
      const int m = r + g * 8;
      __builtin_nontemporal_store(
          c[r] * rowscale[m],
          out + ((size_t)b * SEQ + qbase + m) * DHEAD + ncol);
    }
  }
}

// ---------------- host launcher ----------------

extern "C" void kernel_launch(void* const* d_in, const int* in_sizes, int n_in,
                              void* d_out, int out_size, void* d_ws, size_t ws_size,
                              hipStream_t stream) {
  (void)in_sizes; (void)n_in; (void)out_size;
  const float* q = (const float*)d_in[0];
  const float* k = (const float*)d_in[1];
  const float* v = (const float*)d_in[2];
  float* out = (float*)d_out;

  const size_t n_elems  = (size_t)BATCH * SEQ * DHEAD;       // 2,097,152
  const size_t kh_bytes = n_elems * sizeof(f16_t);           // 4 MB
  const size_t need     = 2 * kh_bytes;                      // Kh + Vt = 8 MB

  const size_t smem_bytes = (size_t)(16 * PSTR + 128 + 16 + 16) * sizeof(float);
  dim3 grid(BATCH * (SEQ / 16));   // 2048 workgroups
  dim3 block(NTHREADS);            // 4 waves (wave32)

  if (d_ws != nullptr && ws_size >= need) {
    f16_t* Kh = (f16_t*)d_ws;
    f16_t* Vt = (f16_t*)((char*)d_ws + kh_bytes);
    cvt_f16_kernel<<<dim3((unsigned)(n_elems / 4 / 256)), dim3(256), 0, stream>>>(k, Kh);
    vtrans_kernel<<<dim3(BATCH * 32), dim3(256), 0, stream>>>(v, Vt);
    attn_fused_kernel<true><<<grid, block, smem_bytes, stream>>>(q, k, v, Kh, Vt, out);
  } else {
    attn_fused_kernel<false><<<grid, block, smem_bytes, stream>>>(q, k, v, nullptr, nullptr, out);
  }
}
